// MultiScaleWaveletAttention_86363202388374
// MI455X (gfx1250) — matrix operand order repaired and checked by reading, not compiled, not run on hardware
//
#include <hip/hip_runtime.h>
#include <hip/hip_bf16.h>

// MultiScaleWaveletAttention for MI455X (gfx1250, wave32).
// Large GEMMs: v_wmma_f32_16x16x32_f16, software-pipelined K loop,
// 32(M)x64(N) tile per wave. DWT row load uses the CDNA5 async-to-LDS path.

#define BATCH  8192
#define DMODEL 512
#define NREG   4
#define NS     6
#define NHID   128
#define NHEADS 4
#define HD     128

typedef __attribute__((ext_vector_type(16))) _Float16 v16h;
typedef __attribute__((ext_vector_type(8)))  _Float16 v8h;
typedef __attribute__((ext_vector_type(4)))  _Float16 v4h;
typedef __attribute__((ext_vector_type(8)))  float    v8f;

__constant__ float c_lo[16] = {
    -0.00011747678400228192f, 0.0006754494059985568f, -0.0003917403729959771f,
    -0.00487035299301066f,    0.008746094047015655f,   0.013981027917015516f,
    -0.04408825393106472f,   -0.01736930100202211f,    0.128747426620186f,
     0.00047248457399797254f,-0.2840155429624281f,    -0.015829105256023893f,
     0.5853546836548691f,     0.6756307362980128f,     0.3128715909144659f,
     0.05441584224308161f};

// ---------------------------------------------------------------- pack kernels
__global__ void pack_f16(const float* __restrict__ in, _Float16* __restrict__ out, int n) {
    int i = blockIdx.x * blockDim.x + threadIdx.x;
    if (i < n) out[i] = (_Float16)in[i];
}

// Wt[n*K + k] = (f16) W[k*N + n], batched over blockIdx.z (stride K*N).
__global__ void pack_transpose(const float* __restrict__ W, _Float16* __restrict__ Wt,
                               int K, int N) {
    size_t boff = (size_t)blockIdx.z * (size_t)K * N;
    int idx = blockIdx.x * blockDim.x + threadIdx.x;
    if (idx >= K * N) return;
    int n = idx / K;
    int k = idx - n * K;
    Wt[boff + (size_t)n * K + k] = (_Float16)W[boff + (size_t)k * N + n];
}

// W1 is (R, D+R, HID). Column col = r*HID + hh.
// W1pT[col*D + d] = W1[r, d, hh]; b1eff[col] = b1[r,hh] + W1[r, D+r, hh] (one-hot row).
__global__ void pack_w1(const float* __restrict__ W1, const float* __restrict__ b1,
                        _Float16* __restrict__ W1pT, float* __restrict__ b1eff) {
    int idx = blockIdx.x * blockDim.x + threadIdx.x;
    if (idx < DMODEL * (NREG * NHID)) {
        int col = idx >> 9;           // / DMODEL
        int d   = idx & (DMODEL - 1);
        int r   = col >> 7;           // / NHID
        int hh  = col & (NHID - 1);
        W1pT[(size_t)col * DMODEL + d] =
            (_Float16)W1[((size_t)r * (DMODEL + NREG) + d) * NHID + hh];
    }
    if (idx < NREG * NHID) {
        int r = idx >> 7, hh = idx & (NHID - 1);
        b1eff[idx] = b1[idx] + W1[((size_t)r * (DMODEL + NREG) + DMODEL + r) * NHID + hh];
    }
}

// ---------------------------------------------------------------- DWT (db8, periodization, 5 levels)
// comps layout: (s, b, d), f16, zero-padded to DMODEL per slot.
// Row load uses GLOBAL_LOAD_ASYNC_TO_LDS_B128 (ASYNCcnt DMA path, ISA 08 §4).
__global__ __launch_bounds__(256) void dwt_kernel(const float* __restrict__ x,
                                                  _Float16* __restrict__ comps) {
    __shared__ float a[DMODEL];
    __shared__ float nxt[DMODEL / 2];
    int b = blockIdx.x;
    int tid = threadIdx.x;

    // Async DMA the 2KB row into LDS: 128 lanes x 16B.
    if (tid < DMODEL / 4) {
        unsigned lds_off = (unsigned)(size_t)(&a[tid * 4]);
        const float* g = x + (size_t)b * DMODEL + tid * 4;
        asm volatile("global_load_async_to_lds_b128 %0, %1, off"
                     :: "v"(lds_off), "v"(g) : "memory");
    }
    // Overlap: zero the padded comps slots while the DMA is in flight.
    for (int i = tid; i < NS * DMODEL; i += 256) {
        int s = i >> 9, d = i & (DMODEL - 1);
        comps[(size_t)s * BATCH * DMODEL + (size_t)b * DMODEL + d] = (_Float16)0.0f;
    }
    asm volatile("s_wait_asynccnt 0" ::: "memory");
    __syncthreads();

    int n = DMODEL;
    for (int lvl = 0; lvl < 5; ++lvl) {
        int half = n >> 1;
        int slot = 5 - lvl;   // d1 -> slot 5 ... d5 -> slot 1
        for (int kk = tid; kk < half; kk += 256) {
            float sa = 0.0f, sd = 0.0f;
            int base = 2 * kk + 1;
            #pragma unroll
            for (int j = 0; j < 16; ++j) {
                int idx = (base - j) & (n - 1);      // n is a power of two
                float av = a[idx];
                float hi = (j & 1) ? c_lo[15 - j] : -c_lo[15 - j];
                sa += c_lo[j] * av;
                sd += hi * av;
            }
            nxt[kk] = sa;
            comps[(size_t)slot * BATCH * DMODEL + (size_t)b * DMODEL + kk] = (_Float16)sd;
        }
        __syncthreads();
        for (int kk = tid; kk < half; kk += 256) a[kk] = nxt[kk];
        __syncthreads();
        n = half;
    }
    if (tid < 16)  // final approximation (len 16) -> slot 0
        comps[(size_t)b * DMODEL + tid] = (_Float16)a[tid];
}

// ---------------------------------------------------------------- WMMA GEMM
// C[M,N] = A[M,K](f16 row-major) * Bt[N,K](f16, B transposed) + bias[N]
// Block: 256 threads = 8 waves; wave tile 32(M) x 64(N) -> block tile 256 x 64.
// K-loop software-pipelined: fragments for k+32 are loaded while the 8 WMMAs
// for k issue, so s_wait covers loads issued a full iteration earlier.
// Fragment layouts, same contiguous K order in A and B (A: row m = lane&15, 16 contiguous K per half;
// B: lane holds column n = lane&15, 16 contiguous K per half).
__device__ __forceinline__ v16h load_a_frag(const _Float16* p) {
    v8h lo = *(const v8h*)(p);
    v8h hi = *(const v8h*)(p + 8);
    return __builtin_shufflevector(lo, hi, 0, 1, 2, 3, 4, 5, 6, 7,
                                   8, 9, 10, 11, 12, 13, 14, 15);
}

template <bool RELU, bool OUT_F16>
__global__ __launch_bounds__(256) void gemm_wmma(
    const _Float16* __restrict__ A, const _Float16* __restrict__ Bt,
    const float* __restrict__ bias, void* __restrict__ Cv,
    int M, int N, int K,
    size_t strideA, size_t strideB, size_t strideBias, size_t strideC) {
    const _Float16* a_p    = A    + (size_t)blockIdx.z * strideA;
    const _Float16* b_p    = Bt   + (size_t)blockIdx.z * strideB;
    const float*    bias_p = bias + (size_t)blockIdx.z * strideBias;

    int lane = threadIdx.x & 31;
    int wave = threadIdx.x >> 5;
    int m0 = (blockIdx.y * 8 + wave) * 32;
    int n0 = blockIdx.x * 64;
    int mr = lane & 15;
    int h8  = (lane >> 4) * 16;   // A chunk select
    int h16 = (lane >> 4) * 16;   // B chunk select

    const _Float16* arow0 = a_p + (size_t)(m0 + mr) * K + h8;
    const _Float16* arow1 = arow0 + (size_t)16 * K;
    const _Float16* br0 = b_p + (size_t)(n0 + mr) * K + h16;
    const _Float16* br1 = br0 + (size_t)16 * K;
    const _Float16* br2 = br0 + (size_t)32 * K;
    const _Float16* br3 = br0 + (size_t)48 * K;

    v8f acc0[4] = {};
    v8f acc1[4] = {};

    v16h a0 = load_a_frag(arow0);
    v16h a1 = load_a_frag(arow1);
    v16h b0 = *(const v16h*)(br0);
    v16h b1 = *(const v16h*)(br1);
    v16h b2 = *(const v16h*)(br2);
    v16h b3 = *(const v16h*)(br3);

    for (int k0 = 32; k0 < K; k0 += 32) {
        v16h na0 = load_a_frag(arow0 + k0);
        v16h na1 = load_a_frag(arow1 + k0);
        v16h nb0 = *(const v16h*)(br0 + k0);
        v16h nb1 = *(const v16h*)(br1 + k0);
        v16h nb2 = *(const v16h*)(br2 + k0);
        v16h nb3 = *(const v16h*)(br3 + k0);

        acc0[0] = __builtin_amdgcn_wmma_f32_16x16x32_f16(false, a0, false, b0, (short)0, acc0[0], false, false);
        acc0[1] = __builtin_amdgcn_wmma_f32_16x16x32_f16(false, a0, false, b1, (short)0, acc0[1], false, false);
        acc0[2] = __builtin_amdgcn_wmma_f32_16x16x32_f16(false, a0, false, b2, (short)0, acc0[2], false, false);
        acc0[3] = __builtin_amdgcn_wmma_f32_16x16x32_f16(false, a0, false, b3, (short)0, acc0[3], false, false);
        acc1[0] = __builtin_amdgcn_wmma_f32_16x16x32_f16(false, a1, false, b0, (short)0, acc1[0], false, false);
        acc1[1] = __builtin_amdgcn_wmma_f32_16x16x32_f16(false, a1, false, b1, (short)0, acc1[1], false, false);
        acc1[2] = __builtin_amdgcn_wmma_f32_16x16x32_f16(false, a1, false, b2, (short)0, acc1[2], false, false);
        acc1[3] = __builtin_amdgcn_wmma_f32_16x16x32_f16(false, a1, false, b3, (short)0, acc1[3], false, false);

        a0 = na0; a1 = na1; b0 = nb0; b1 = nb1; b2 = nb2; b3 = nb3;
    }
    acc0[0] = __builtin_amdgcn_wmma_f32_16x16x32_f16(false, a0, false, b0, (short)0, acc0[0], false, false);
    acc0[1] = __builtin_amdgcn_wmma_f32_16x16x32_f16(false, a0, false, b1, (short)0, acc0[1], false, false);
    acc0[2] = __builtin_amdgcn_wmma_f32_16x16x32_f16(false, a0, false, b2, (short)0, acc0[2], false, false);
    acc0[3] = __builtin_amdgcn_wmma_f32_16x16x32_f16(false, a0, false, b3, (short)0, acc0[3], false, false);
    acc1[0] = __builtin_amdgcn_wmma_f32_16x16x32_f16(false, a1, false, b0, (short)0, acc1[0], false, false);
    acc1[1] = __builtin_amdgcn_wmma_f32_16x16x32_f16(false, a1, false, b1, (short)0, acc1[1], false, false);
    acc1[2] = __builtin_amdgcn_wmma_f32_16x16x32_f16(false, a1, false, b2, (short)0, acc1[2], false, false);
    acc1[3] = __builtin_amdgcn_wmma_f32_16x16x32_f16(false, a1, false, b3, (short)0, acc1[3], false, false);

    // C/D layout: lane<16 -> N=lane, M=v; lane>=16 -> N=lane-16, M=8+v.
    int c  = lane & 15;
    int rb = (lane >> 4) * 8;
    float*    Cf = ((float*)Cv)    + (size_t)blockIdx.z * strideC;
    _Float16* Ch = ((_Float16*)Cv) + (size_t)blockIdx.z * strideC;
    #pragma unroll
    for (int j = 0; j < 4; ++j) {
        int n = n0 + j * 16 + c;
        float bz = bias_p[n];
        #pragma unroll
        for (int v = 0; v < 8; ++v) {
            float v0 = acc0[j][v] + bz;
            float v1 = acc1[j][v] + bz;
            if (RELU) { v0 = fmaxf(v0, 0.0f); v1 = fmaxf(v1, 0.0f); }
            size_t off0 = (size_t)(m0 + rb + v) * N + n;
            size_t off1 = (size_t)(m0 + 16 + rb + v) * N + n;
            if constexpr (OUT_F16) { Ch[off0] = (_Float16)v0; Ch[off1] = (_Float16)v1; }
            else                   { Cf[off0] = v0;           Cf[off1] = v1; }
        }
    }
}

// ---------------------------------------------------------------- gating
// Only the r = labels[b] slice of logits is needed.
__global__ void gate_kernel(const float* __restrict__ h, const float* __restrict__ W2,
                            const float* __restrict__ b2, const int* __restrict__ labels,
                            float* __restrict__ w) {
    int b = blockIdx.x * blockDim.x + threadIdx.x;
    if (b >= BATCH) return;
    int r = labels[b];
    float lg[NS];
    #pragma unroll
    for (int s = 0; s < NS; ++s) lg[s] = b2[r * NS + s];
    const float* hr  = h  + (size_t)b * (NREG * NHID) + (size_t)r * NHID;
    const float* w2r = W2 + (size_t)r * NHID * NS;
    for (int hh = 0; hh < NHID; ++hh) {
        float hv = hr[hh];
        #pragma unroll
        for (int s = 0; s < NS; ++s) lg[s] += hv * w2r[hh * NS + s];
    }
    float mx = lg[0];
    #pragma unroll
    for (int s = 1; s < NS; ++s) mx = fmaxf(mx, lg[s]);
    float den = 0.0f, e[NS];
    #pragma unroll
    for (int s = 0; s < NS; ++s) { e[s] = expf(lg[s] - mx); den += e[s]; }
    float inv = 1.0f / den;
    #pragma unroll
    for (int s = 0; s < NS; ++s) w[(size_t)b * NS + s] = e[s] * inv;
}

// ---------------------------------------------------------------- attention (seq len 6)
// One wave per (b, head): lane owns 4 contiguous d's; 6x6 scores via shfl reductions.
__global__ __launch_bounds__(256) void attn_kernel(const _Float16* __restrict__ Q,
                                                   const _Float16* __restrict__ Kb,
                                                   const _Float16* __restrict__ V,
                                                   _Float16* __restrict__ Ctx) {
    int wave = threadIdx.x >> 5;
    int lane = threadIdx.x & 31;
    int b  = blockIdx.x * 2 + (wave >> 2);
    int hh = wave & 3;
    int d0 = hh * HD + lane * 4;

    float q6[NS][4], k6[NS][4], v6[NS][4];
    #pragma unroll
    for (int s = 0; s < NS; ++s) {
        size_t row = ((size_t)s * BATCH + b) * DMODEL + d0;
        v4h qv = *(const v4h*)(Q + row);
        v4h kv = *(const v4h*)(Kb + row);
        v4h vv = *(const v4h*)(V + row);
        #pragma unroll
        for (int t = 0; t < 4; ++t) {
            q6[s][t] = (float)qv[t]; k6[s][t] = (float)kv[t]; v6[s][t] = (float)vv[t];
        }
    }

    float att[NS][NS];
    #pragma unroll
    for (int qi = 0; qi < NS; ++qi)
        #pragma unroll
        for (int ki = 0; ki < NS; ++ki) {
            float p = q6[qi][0] * k6[ki][0] + q6[qi][1] * k6[ki][1] +
                      q6[qi][2] * k6[ki][2] + q6[qi][3] * k6[ki][3];
            #pragma unroll
            for (int off = 16; off > 0; off >>= 1) p += __shfl_xor(p, off, 32);
            att[qi][ki] = p * 0.08838834764831845f;   // 1/sqrt(128)
        }

    #pragma unroll
    for (int qi = 0; qi < NS; ++qi) {
        float mx = att[qi][0];
        #pragma unroll
        for (int ki = 1; ki < NS; ++ki) mx = fmaxf(mx, att[qi][ki]);
        float den = 0.0f;
        #pragma unroll
        for (int ki = 0; ki < NS; ++ki) { att[qi][ki] = expf(att[qi][ki] - mx); den += att[qi][ki]; }
        float inv = 1.0f / den;
        float o[4] = {0.f, 0.f, 0.f, 0.f};
        #pragma unroll
        for (int ki = 0; ki < NS; ++ki) {
            float aw = att[qi][ki] * inv;
            #pragma unroll
            for (int t = 0; t < 4; ++t) o[t] += aw * v6[ki][t];
        }
        v4h ov;
        #pragma unroll
        for (int t = 0; t < 4; ++t) ov[t] = (_Float16)o[t];
        *(v4h*)(Ctx + ((size_t)qi * BATCH + b) * DMODEL + d0) = ov;
    }
}

// ---------------------------------------------------------------- weighted sum + residual + LN
__global__ __launch_bounds__(256) void final_kernel(const float* __restrict__ x,
                                                    const _Float16* __restrict__ refined,
                                                    const float* __restrict__ w,
                                                    const float* __restrict__ ln_g,
                                                    const float* __restrict__ ln_b,
                                                    float* __restrict__ out) {
    int b = blockIdx.x;
    int tid = threadIdx.x;
    float wv[NS];
    #pragma unroll
    for (int s = 0; s < NS; ++s) wv[s] = w[(size_t)b * NS + s];

    float y[2];
    #pragma unroll
    for (int e = 0; e < 2; ++e) {
        int d = tid + e * 256;
        float acc = x[(size_t)b * DMODEL + d];
        #pragma unroll
        for (int s = 0; s < NS; ++s)
            acc += wv[s] * (float)refined[((size_t)s * BATCH + b) * DMODEL + d];
        y[e] = acc;
    }
    float s1 = y[0] + y[1];
    float s2 = y[0] * y[0] + y[1] * y[1];
    #pragma unroll
    for (int off = 16; off > 0; off >>= 1) {
        s1 += __shfl_xor(s1, off, 32);
        s2 += __shfl_xor(s2, off, 32);
    }
    __shared__ float rs1[8], rs2[8];
    int wave = tid >> 5, lane = tid & 31;
    if (lane == 0) { rs1[wave] = s1; rs2[wave] = s2; }
    __syncthreads();
    float t1 = 0.0f, t2 = 0.0f;
    #pragma unroll
    for (int i = 0; i < 8; ++i) { t1 += rs1[i]; t2 += rs2[i]; }
    float mean = t1 * (1.0f / DMODEL);
    float var  = t2 * (1.0f / DMODEL) - mean * mean;
    float rstd = rsqrtf(var + 1e-5f);
    #pragma unroll
    for (int e = 0; e < 2; ++e) {
        int d = tid + e * 256;
        out[(size_t)b * DMODEL + d] = (y[e] - mean) * rstd * ln_g[d] + ln_b[d];
    }
}

// ---------------------------------------------------------------- launch
extern "C" void kernel_launch(void* const* d_in, const int* in_sizes, int n_in,
                              void* d_out, int out_size, void* d_ws, size_t ws_size,
                              hipStream_t stream) {
    (void)in_sizes; (void)n_in; (void)out_size; (void)ws_size;
    const float* x   = (const float*)d_in[0];
    const float* W1  = (const float*)d_in[1];
    const float* b1  = (const float*)d_in[2];
    const float* W2  = (const float*)d_in[3];
    const float* b2  = (const float*)d_in[4];
    const float* Wp  = (const float*)d_in[5];
    const float* bp  = (const float*)d_in[6];
    const float* Wq  = (const float*)d_in[7];
    const float* bq  = (const float*)d_in[8];
    const float* Wk  = (const float*)d_in[9];
    const float* bk  = (const float*)d_in[10];
    const float* Wv  = (const float*)d_in[11];
    const float* bv  = (const float*)d_in[12];
    const float* Wo  = (const float*)d_in[13];
    const float* bo  = (const float*)d_in[14];
    const float* lng = (const float*)d_in[15];
    const float* lnb = (const float*)d_in[16];
    const int*   lbl = (const int*)d_in[17];
    float* out = (float*)d_out;

    char* ws = (char*)d_ws;
    size_t off = 0;
    auto carve = [&](size_t bytes) -> char* {
        char* p = ws + off;
        off += (bytes + 255) & ~(size_t)255;
        return p;
    };
    const size_t SBD = (size_t)NS * BATCH * DMODEL;
    _Float16* Xh    = (_Float16*)carve((size_t)BATCH * DMODEL * 2);
    _Float16* COMPS = (_Float16*)carve(SBD * 2);   // comps, later attention ctx
    float*    Hbuf  = (float*)   carve((size_t)BATCH * NREG * NHID * 4);
    float*    WG    = (float*)   carve((size_t)BATCH * NS * 4);
    _Float16* Pbuf  = (_Float16*)carve(SBD * 2);   // comps_p
    _Float16* Qb    = (_Float16*)carve(SBD * 2);   // q, later refined
    _Float16* Kbuf  = (_Float16*)carve(SBD * 2);
    _Float16* Vbuf  = (_Float16*)carve(SBD * 2);
    _Float16* W1PT  = (_Float16*)carve((size_t)DMODEL * DMODEL * 2);
    float*    B1E   = (float*)   carve((size_t)DMODEL * 4);
    _Float16* WPT   = (_Float16*)carve((size_t)NS * DMODEL * DMODEL * 2);
    _Float16* WQT   = (_Float16*)carve((size_t)DMODEL * DMODEL * 2);
    _Float16* WKT   = (_Float16*)carve((size_t)DMODEL * DMODEL * 2);
    _Float16* WVT   = (_Float16*)carve((size_t)DMODEL * DMODEL * 2);
    _Float16* WOT   = (_Float16*)carve((size_t)DMODEL * DMODEL * 2);

    const int TP = 256;
    int nDD = DMODEL * DMODEL;

    pack_f16<<<(BATCH * DMODEL + TP - 1) / TP, TP, 0, stream>>>(x, Xh, BATCH * DMODEL);
    pack_w1<<<(nDD + TP - 1) / TP, TP, 0, stream>>>(W1, b1, W1PT, B1E);
    pack_transpose<<<dim3((nDD + TP - 1) / TP, 1, NS), TP, 0, stream>>>(Wp, WPT, DMODEL, DMODEL);
    pack_transpose<<<dim3((nDD + TP - 1) / TP, 1, 1), TP, 0, stream>>>(Wq, WQT, DMODEL, DMODEL);
    pack_transpose<<<dim3((nDD + TP - 1) / TP, 1, 1), TP, 0, stream>>>(Wk, WKT, DMODEL, DMODEL);
    pack_transpose<<<dim3((nDD + TP - 1) / TP, 1, 1), TP, 0, stream>>>(Wv, WVT, DMODEL, DMODEL);
    pack_transpose<<<dim3((nDD + TP - 1) / TP, 1, 1), TP, 0, stream>>>(Wo, WOT, DMODEL, DMODEL);

    dwt_kernel<<<BATCH, TP, 0, stream>>>(x, COMPS);

    // gating: h = relu(x @ W1 + b1eff), f32 out. Block tile 256(M) x 64(N).
    gemm_wmma<true, false><<<dim3(DMODEL / 64, BATCH / 256, 1), TP, 0, stream>>>(
        Xh, W1PT, B1E, Hbuf, BATCH, DMODEL, DMODEL, 0, 0, 0, 0);
    gate_kernel<<<BATCH / TP, TP, 0, stream>>>(Hbuf, W2, b2, lbl, WG);

    // comps_p[s] = comps[s] @ Wp[s] + bp[s]   (grid.z = s)
    gemm_wmma<false, true><<<dim3(DMODEL / 64, BATCH / 256, NS), TP, 0, stream>>>(
        COMPS, WPT, bp, Pbuf, BATCH, DMODEL, DMODEL,
        (size_t)BATCH * DMODEL, (size_t)DMODEL * DMODEL, DMODEL, (size_t)BATCH * DMODEL);

    // q/k/v over M = S*B rows
    gemm_wmma<false, true><<<dim3(DMODEL / 64, (NS * BATCH) / 256, 1), TP, 0, stream>>>(
        Pbuf, WQT, bq, Qb, NS * BATCH, DMODEL, DMODEL, 0, 0, 0, 0);
    gemm_wmma<false, true><<<dim3(DMODEL / 64, (NS * BATCH) / 256, 1), TP, 0, stream>>>(
        Pbuf, WKT, bk, Kbuf, NS * BATCH, DMODEL, DMODEL, 0, 0, 0, 0);
    gemm_wmma<false, true><<<dim3(DMODEL / 64, (NS * BATCH) / 256, 1), TP, 0, stream>>>(
        Pbuf, WVT, bv, Vbuf, NS * BATCH, DMODEL, DMODEL, 0, 0, 0, 0);

    attn_kernel<<<BATCH / 2, TP, 0, stream>>>(Qb, Kbuf, Vbuf, COMPS);

    // refined = ctx @ Wo + bo  (into Qb, q no longer needed)
    gemm_wmma<false, true><<<dim3(DMODEL / 64, (NS * BATCH) / 256, 1), TP, 0, stream>>>(
        COMPS, WOT, bo, Qb, NS * BATCH, DMODEL, DMODEL, 0, 0, 0, 0);

    final_kernel<<<BATCH, TP, 0, stream>>>(x, Qb, WG, lng, lnb, out);
}
